// attention_44367012167884
// MI455X (gfx1250) — matrix operand order repaired and checked by reading, not compiled, not run on hardware
//
#include <hip/hip_runtime.h>
#include <hip/hip_bf16.h>

// ============================================================================
// MI455X (gfx1250, wave32) dual-softmax attention.
//
// Pipeline:
//   1) proj_gemm_bf16  x3 : P = X @ W^T + b  (fp32 in, bf16 out, f32 WMMA acc)
//                           stored as [B, H, L, d_k] bf16 in workspace.
//   2) dual_attn           : per (b,h,qtile) flash-style ONLINE dual softmax,
//                            acc1 = sum exp(QK^T/8 - m1) (K+V),
//                            acc2 = sum exp(QV^T/8 - m2) (K+V),
//                            temp = acc1/l1 + acc2/l2, partial = sum_q temp*Q.
//                            Q tile staged via TENSOR_LOAD_TO_LDS (TDM).
//   3) reduce_logits       : deterministic sum of 16 q-tile partials -> d_out.
//
// All matrix math uses v_wmma_f32_16x16x32_bf16; K+V uses v_pk_add_bf16;
// Q staging uses the Tensor Data Mover (tensor_load_to_lds + s_wait_tensorcnt).
// Attention maps never touch HBM (would be 537 MB each); traffic ~250 MB.
//
// LDS strides are chosen so every 16-lane fragment load (16 B per lane,
// lane-stride == row-stride) tiles all 64 LDS banks exactly:
//   stride 40 shorts (80 B  -> 20 banks/lane), 88 (176 B -> 44 banks/lane),
//   136 (272 B -> 68==4 mod 64 banks/lane); all multiples of 16 B so the
//   compiler can merge fragment gathers into ds_load_b128.
// ============================================================================

typedef __attribute__((ext_vector_type(16))) __bf16 v16bf;
typedef __attribute__((ext_vector_type(8)))  float  v8f;
typedef __attribute__((ext_vector_type(4)))  unsigned int u32x4;
typedef __attribute__((ext_vector_type(8)))  int          i32x8;
typedef __attribute__((ext_vector_type(4)))  int          i32x4;

union Frag {
  v16bf v;
  unsigned int u[8];
};

#define NHEAD 16
#define NHID  1024
#define NB    8
#define NL    1024
#define DKH   64

// LDS row strides (in 16-bit elements) -- see banking note above.
#define STRIDE_G 40   // proj GEMM A/B tiles (32 K columns + pad)
#define STRIDE_Q 88   // Q/K/V tiles (64 columns + pad)
#define STRIDE_T 136  // X^T and P tiles (128 columns + pad)

__device__ __forceinline__ unsigned short f2bf(float x) {
  unsigned int u = __builtin_bit_cast(unsigned int, x);
  return (unsigned short)((u + 0x7FFFu + ((u >> 16) & 1u)) >> 16);  // RNE
}
__device__ __forceinline__ float bf2f(unsigned short h) {
  unsigned int u = ((unsigned int)h) << 16;
  return __builtin_bit_cast(float, u);
}
// Packed bf16 add: one VOP3P op instead of ~12 scalar VALU ops.
__device__ __forceinline__ unsigned int pk_add_bf16(unsigned int a, unsigned int b) {
  unsigned int r;
  asm("v_pk_add_bf16 %0, %1, %2" : "=v"(r) : "v"(a), "v"(b));
  return r;
}

// A fragment 16x32 bf16 (M x K). Per CDNA5 ISA: lanes 0-15 (M=lane) hold
// K {0..7, 16..23}; lanes 16-31 hold K {8..15, 24..31}. Two contiguous 16 B
// chunks per lane -> compiler merges into 2x ds_load_b128.
__device__ __forceinline__ Frag load_a_frag(const unsigned short* lds, int row0,
                                            int col0, int stride) {
  int lane = threadIdx.x & 31;
  int lm = lane & 15, half = lane >> 4;
  const unsigned short* p = lds + (row0 + lm) * stride + col0 + half * 8;
  Frag f;
#pragma unroll
  for (int jp = 0; jp < 8; ++jp) {
    int kk = (jp < 4) ? (2 * jp) : (2 * jp + 8);   // pairs {0,2,4,6,16,18,20,22}
    f.u[jp] = *(const unsigned int*)(p + kk);
  }
  return f;
}

// B fragment 32x16 bf16 (K x N), LDS stores one row per N index, contiguous
// in K: B[k][n] = lds[(nrow0+n)*stride + col0 + k]. lane%16 = N; lanes 0-15
// hold K {0..7, 16..23}, lanes 16-31 hold K {8..15, 24..31} (matches A).
__device__ __forceinline__ Frag load_b_rows(const unsigned short* lds, int nrow0,
                                            int col0, int stride) {
  int lane = threadIdx.x & 31;
  int lm = lane & 15, half = lane >> 4;
  const unsigned short* p = lds + (nrow0 + lm) * stride + col0 + half * 8;
  Frag f;
#pragma unroll
  for (int jp = 0; jp < 8; ++jp) f.u[jp] = *(const unsigned int*)(p + ((jp < 4) ? (2 * jp) : (2 * jp + 8)));
  return f;
}

__device__ __forceinline__ v8f wmma_bf16(Frag a, Frag b, v8f c) {
  return __builtin_amdgcn_wmma_f32_16x16x32_bf16(false, a.v, false, b.v,
                                                 (short)0, c, false, false);
}

// ============================================================================
// Kernel 1: P = X @ W^T + bias, stored bf16 as [B, H, L, DKH].
// Grid (64, 8); 256 threads = 8 waves, WG tile 128(M) x 128(N), K step 32.
// ============================================================================
__global__ __launch_bounds__(256) void proj_gemm_bf16(
    const float* __restrict__ X, const float* __restrict__ W,
    const float* __restrict__ bias, unsigned short* __restrict__ out) {
  __shared__ unsigned short lsA[128 * STRIDE_G];
  __shared__ unsigned short lsB[128 * STRIDE_G];
  const int t = threadIdx.x;
  const int wave = t >> 5;
  const int wm = wave & 1;   // 2 M-groups of 64 rows
  const int wn = wave >> 1;  // 4 N-groups of 32 cols
  const int gm0 = blockIdx.x * 128;
  const int gn0 = blockIdx.y * 128;
  const int lane = t & 31;
  const int lm = lane & 15, half = lane >> 4;

  v8f acc[4][2];
#pragma unroll
  for (int i = 0; i < 4; ++i)
#pragma unroll
    for (int j = 0; j < 2; ++j) acc[i][j] = {};

  const int row = t >> 1;
  const int col0 = (t & 1) * 16;

  for (int kk0 = 0; kk0 < NHID; kk0 += 32) {
    __syncthreads();
    // Stage A (activations) and B (weight rows; both K-contiguous) as bf16.
#pragma unroll
    for (int i = 0; i < 4; ++i) {
      float4 fa = *(const float4*)(X + (gm0 + row) * NHID + kk0 + col0 + i * 4);
      float4 fb = *(const float4*)(W + (gn0 + row) * NHID + kk0 + col0 + i * 4);
      *(unsigned int*)&lsA[row * STRIDE_G + col0 + i * 4 + 0] =
          (unsigned int)f2bf(fa.x) | ((unsigned int)f2bf(fa.y) << 16);
      *(unsigned int*)&lsA[row * STRIDE_G + col0 + i * 4 + 2] =
          (unsigned int)f2bf(fa.z) | ((unsigned int)f2bf(fa.w) << 16);
      *(unsigned int*)&lsB[row * STRIDE_G + col0 + i * 4 + 0] =
          (unsigned int)f2bf(fb.x) | ((unsigned int)f2bf(fb.y) << 16);
      *(unsigned int*)&lsB[row * STRIDE_G + col0 + i * 4 + 2] =
          (unsigned int)f2bf(fb.z) | ((unsigned int)f2bf(fb.w) << 16);
    }
    if (kk0 + 32 < NHID) {  // global_prefetch_b8 of the next K slab
      __builtin_prefetch(X + (gm0 + row) * NHID + kk0 + 32 + col0, 0, 0);
      __builtin_prefetch(W + (gn0 + row) * NHID + kk0 + 32 + col0, 0, 0);
    }
    __syncthreads();

    Frag aF[4], bF[2];
#pragma unroll
    for (int i = 0; i < 4; ++i)
      aF[i] = load_a_frag(lsA, wm * 64 + i * 16, 0, STRIDE_G);
#pragma unroll
    for (int j = 0; j < 2; ++j)
      bF[j] = load_b_rows(lsB, wn * 32 + j * 16, 0, STRIDE_G);
#pragma unroll
    for (int i = 0; i < 4; ++i)
#pragma unroll
      for (int j = 0; j < 2; ++j) acc[i][j] = wmma_bf16(aF[i], bF[j], acc[i][j]);
  }

  // Epilogue: + bias, store bf16 head-split [B, H, L, DKH].
  // C layout: VGPR r, lane: M = r + 8*(lane/16), N = lane%16.
#pragma unroll
  for (int j = 0; j < 2; ++j) {
    int gn = gn0 + wn * 32 + j * 16 + lm;
    float bj = bias[gn];
    int hh = gn >> 6, d = gn & 63;
#pragma unroll
    for (int i = 0; i < 4; ++i) {
#pragma unroll
      for (int r = 0; r < 8; ++r) {
        int gm = gm0 + wm * 64 + i * 16 + r + 8 * half;
        int bb = gm >> 10, lrow = gm & (NL - 1);
        out[(((bb * NHEAD + hh) * NL) + lrow) * DKH + d] = f2bf(acc[i][j][r] + bj);
      }
    }
  }
}

// ============================================================================
// Kernel 2: dual online-softmax attention.
// Grid (16 qtiles, 16 heads, 8 batch); 128 threads = 4 waves; each wave owns
// 16 q rows and the full kv width (row stats stay wave-local).
// LDS partition (bytes, all 16B-aligned; sQ MUST stay at offset 0 -- the TDM
// descriptor addresses it as LDS address 0):
//   sQ  64xSTRIDE_Q   = 11264
//   sK  128xSTRIDE_Q  = 22528
//   sV  128xSTRIDE_Q  = 22528
//   sXT 64xSTRIDE_T   = 17408   (X = K+V, stored TRANSPOSED: [d][kv])
//   sP  4 x 16xSTRIDE_T = 17408 (per-wave P transpose scratch)
//   sRed 4x64 floats  = 1024
// ============================================================================
#define OFF_K   11264
#define OFF_V   (OFF_K + 22528)
#define OFF_XT  (OFF_V + 22528)
#define OFF_P   (OFF_XT + 17408)
#define OFF_RED (OFF_P + 17408)
#define ATT_SMEM (OFF_RED + 1024)

__global__ __launch_bounds__(128) void dual_attn(
    const unsigned short* __restrict__ Qp, const unsigned short* __restrict__ Kp,
    const unsigned short* __restrict__ Vp, float* __restrict__ part) {
  extern __shared__ char smem[];
  unsigned short* sQ  = (unsigned short*)(smem);
  unsigned short* sK  = (unsigned short*)(smem + OFF_K);
  unsigned short* sV  = (unsigned short*)(smem + OFF_V);
  unsigned short* sXT = (unsigned short*)(smem + OFF_XT);
  unsigned short* sP  = (unsigned short*)(smem + OFF_P);
  float* sRed = (float*)(smem + OFF_RED);

  const int t = threadIdx.x;
  const int wave = t >> 5;
  const int lane = t & 31;
  const int lm = lane & 15, half = lane >> 4;
  const int qt = blockIdx.x, hh = blockIdx.y, bb = blockIdx.z;

  const long base = ((long)(bb * NHEAD + hh)) * NL * DKH;

  // ---- Stage the 64x64 bf16 Q tile ------------------------------------
  // NOTE: Q is staged UNPADDED (row stride 64 shorts) when the TDM path is
  // used: the copy is a flat 4096-element 1-D tile starting at LDS addr 0.
#if defined(__AMDGCN__) && __has_builtin(__builtin_amdgcn_tensor_load_to_lds) && \
    __has_builtin(__builtin_amdgcn_s_wait_tensorcnt)
#define Q_STRIDE 64
  if (t < 32) {  // one TDM op, issued by wave 0 only
    unsigned long long ga =
        (unsigned long long)(Qp + base + (long)qt * 64 * DKH);
    // D# group 0: count=1 | lds_addr=0 | global_addr[56:0] | type=2
    u32x4 g0 = {1u, 0u, (unsigned int)ga,
                (unsigned int)(ga >> 32) | (2u << 30)};
    // D# group 1: 1-D copy of 4096 2-byte elements.
    //  w0: workgroup_mask=0, data_size=1 (2B)        -> bits [17:16]
    //  w1[31:16] = tensor_dim0[15:0]  (4096)
    //  w3[31:16] = tile_dim0          (4096)
    //  w5        = tensor_dim0_stride (4096)
    i32x8 g1 = {(int)(1u << 16), (int)(4096u << 16), 0, (int)(4096u << 16),
                0, 4096, 0, 0};
    i32x4 z4 = {0, 0, 0, 0};
    i32x8 z8 = {0, 0, 0, 0, 0, 0, 0, 0};
    __builtin_amdgcn_tensor_load_to_lds(g0, g1, z4, z4, z8, 0);
    __builtin_amdgcn_s_wait_tensorcnt(0);
  }
#else
#define Q_STRIDE STRIDE_Q
  {
    const uint4* g = (const uint4*)(Qp + base + (long)qt * 64 * DKH);
#pragma unroll
    for (int i = 0; i < 4; ++i) {
      int flat = t + i * 128;          // uint4 index: row*8 + chunk
      int row = flat >> 3, c8 = flat & 7;
      *((uint4*)(sQ + row * Q_STRIDE) + c8) = g[flat];
    }
  }
#endif
  __syncthreads();

  Frag aQ[2];
#pragma unroll
  for (int c = 0; c < 2; ++c) aQ[c] = load_a_frag(sQ, wave * 16, c * 32, Q_STRIDE);

  v8f a1[4] = {}, a2[4] = {};
  float m1[8], l1[8], m2[8], l2[8];
#pragma unroll
  for (int r = 0; r < 8; ++r) {
    m1[r] = -1e30f; l1[r] = 0.f; m2[r] = -1e30f; l2[r] = 0.f;
  }

  unsigned short* sPw = sP + wave * 16 * STRIDE_T;

  for (int it = 0; it < 8; ++it) {
    __syncthreads();  // all waves done reading previous sK/sV/sXT
    {
      const uint4* gk = (const uint4*)(Kp + base + (long)it * 128 * DKH);
      const uint4* gv = (const uint4*)(Vp + base + (long)it * 128 * DKH);
#pragma unroll
      for (int i = 0; i < 8; ++i) {
        int flat = t + i * 128;        // uint4 index: kvrow*8 + chunk
        int row = flat >> 3, c8 = flat & 7;
        uint4 kq = gk[flat];
        uint4 vq = gv[flat];
        *((uint4*)(sK + row * STRIDE_Q) + c8) = kq;
        *((uint4*)(sV + row * STRIDE_Q) + c8) = vq;
        // X = K + V (packed bf16 add), scattered transposed into sXT[d][kv].
        const unsigned int* kp = (const unsigned int*)&kq;
        const unsigned int* vp = (const unsigned int*)&vq;
        int d0 = c8 * 8;
#pragma unroll
        for (int e = 0; e < 4; ++e) {
          unsigned int x = pk_add_bf16(kp[e], vp[e]);
          sXT[(d0 + 2 * e + 0) * STRIDE_T + row] = (unsigned short)(x & 0xFFFFu);
          sXT[(d0 + 2 * e + 1) * STRIDE_T + row] = (unsigned short)(x >> 16);
        }
      }
    }
    __syncthreads();

    auto do_map = [&](const unsigned short* sS, float* m, float* l, v8f* acc) {
      // S[16 q, 128 kv] = Qtile @ sS^T * 1/sqrt(64)
      v8f S[8];
#pragma unroll
      for (int nt = 0; nt < 8; ++nt) {
        v8f s = {};
#pragma unroll
        for (int c = 0; c < 2; ++c)
          s = wmma_bf16(aQ[c], load_b_rows(sS, nt * 16, c * 32, STRIDE_Q), s);
#pragma unroll
        for (int r = 0; r < 8; ++r) s[r] *= 0.125f;
        S[nt] = s;
      }
      // Online softmax; row M = r + 8*half lives in one 16-lane group, so
      // xor-shuffles with masks 1/2/4/8 stay inside the row's half.
#pragma unroll
      for (int r = 0; r < 8; ++r) {
        float tm = S[0][r];
#pragma unroll
        for (int nt = 1; nt < 8; ++nt) tm = fmaxf(tm, S[nt][r]);
        tm = fmaxf(tm, __shfl_xor(tm, 1));
        tm = fmaxf(tm, __shfl_xor(tm, 2));
        tm = fmaxf(tm, __shfl_xor(tm, 4));
        tm = fmaxf(tm, __shfl_xor(tm, 8));
        float mnew = fmaxf(m[r], tm);
        float fs = __expf(m[r] - mnew);
#pragma unroll
        for (int tt = 0; tt < 4; ++tt) acc[tt][r] *= fs;  // rescale running acc
        float rs = 0.f;
#pragma unroll
        for (int nt = 0; nt < 8; ++nt) {
          float e = __expf(S[nt][r] - mnew);
          S[nt][r] = e;
          rs += e;
        }
        rs += __shfl_xor(rs, 1);
        rs += __shfl_xor(rs, 2);
        rs += __shfl_xor(rs, 4);
        rs += __shfl_xor(rs, 8);
        l[r] = l[r] * fs + rs;
        m[r] = mnew;
      }
      // Transpose C-layout P into A-fragment layout via per-wave LDS scratch.
#pragma unroll
      for (int nt = 0; nt < 8; ++nt)
#pragma unroll
        for (int r = 0; r < 8; ++r)
          sPw[(r + 8 * half) * STRIDE_T + nt * 16 + lm] = f2bf(S[nt][r]);
      // acc[16 q, 64 d] += P @ X. X is transposed in LDS, so the B operand is
      // the fast contiguous-K path (2x ds_load_b128 per fragment).
#pragma unroll
      for (int c = 0; c < 4; ++c) {
        Frag aP = load_a_frag(sPw, 0, c * 32, STRIDE_T);
#pragma unroll
        for (int dt = 0; dt < 4; ++dt)
          acc[dt] = wmma_bf16(aP, load_b_rows(sXT, dt * 16, c * 32, STRIDE_T),
                              acc[dt]);
      }
    };
    do_map(sK, m1, l1, a1);  // att1 = softmax(Q K^T / 8)
    do_map(sV, m2, l2, a2);  // att2 = softmax(Q V^T / 8)
  }

  // temp = acc1/l1 + acc2/l2; partial[d] = sum over this wave's q rows of
  // temp[q,d] * Q[q,d]; then reduce across halves and waves.
  float pr[4] = {0.f, 0.f, 0.f, 0.f};
#pragma unroll
  for (int r = 0; r < 8; ++r) {
    float i1 = 1.f / l1[r];
    float i2 = 1.f / l2[r];
    int row = wave * 16 + r + 8 * half;
#pragma unroll
    for (int tt = 0; tt < 4; ++tt) {
      float tv = a1[tt][r] * i1 + a2[tt][r] * i2;
      float qv = bf2f(sQ[row * Q_STRIDE + tt * 16 + lm]);
      pr[tt] += tv * qv;
    }
  }
#pragma unroll
  for (int tt = 0; tt < 4; ++tt) pr[tt] += __shfl_xor(pr[tt], 16);
  if (half == 0) {
#pragma unroll
    for (int tt = 0; tt < 4; ++tt) sRed[wave * 64 + tt * 16 + lm] = pr[tt];
  }
  __syncthreads();
  if (t < 64) {
    float s = sRed[t] + sRed[64 + t] + sRed[128 + t] + sRed[192 + t];
    part[(((qt * NB) + bb) * NHEAD + hh) * DKH + t] = s;  // [qt][b][h][d]
  }
}

// ============================================================================
// Kernel 3: deterministic reduction of 16 q-tile partials into d_out.
// part inner layout [b][h][d] flat == output flat [b][h*64+d].
// ============================================================================
__global__ __launch_bounds__(256) void reduce_logits(const float* __restrict__ part,
                                                     float* __restrict__ out) {
  int i = blockIdx.x * 256 + threadIdx.x;
  if (i < NB * NHID) {
    float s = 0.f;
#pragma unroll
    for (int qt = 0; qt < 16; ++qt) s += part[qt * (NB * NHID) + i];
    out[i] = s;
  }
}

// ============================================================================
extern "C" void kernel_launch(void* const* d_in, const int* in_sizes, int n_in,
                              void* d_out, int out_size, void* d_ws, size_t ws_size,
                              hipStream_t stream) {
  (void)in_sizes; (void)n_in; (void)out_size; (void)ws_size;
  const float* v  = (const float*)d_in[0];
  const float* k  = (const float*)d_in[1];
  const float* q  = (const float*)d_in[2];
  const float* Wv = (const float*)d_in[3];
  const float* bv = (const float*)d_in[4];
  const float* Wk = (const float*)d_in[5];
  const float* bk = (const float*)d_in[6];
  const float* Wq = (const float*)d_in[7];
  const float* bq = (const float*)d_in[8];

  const size_t projElems = (size_t)NB * NL * NHID;  // 8,388,608 bf16 each
  unsigned short* Vp = (unsigned short*)d_ws;
  unsigned short* Kp = Vp + projElems;
  unsigned short* Qp = Kp + projElems;
  float* part = (float*)(Qp + projElems);  // 16 * 8192 floats

  dim3 gGemm(NB * NL / 128, NHID / 128);
  proj_gemm_bf16<<<gGemm, 256, 0, stream>>>(v, Wv, bv, Vp);
  proj_gemm_bf16<<<gGemm, 256, 0, stream>>>(k, Wk, bk, Kp);
  proj_gemm_bf16<<<gGemm, 256, 0, stream>>>(q, Wq, bq, Qp);

  dim3 gAttn(NL / 64, NHEAD, NB);
  dual_attn<<<gAttn, 128, ATT_SMEM, stream>>>(Qp, Kp, Vp, part);

  reduce_logits<<<(NB * NHID + 255) / 256, 256, 0, stream>>>(part, (float*)d_out);
}
